// CausalSelfAttention_20495583936773
// MI455X (gfx1250) — matrix operand order repaired and checked by reading, not compiled, not run on hardware
//
#include <hip/hip_runtime.h>
#include <hip/hip_bf16.h>
#include <stdint.h>

// Problem constants (match reference)
#define CB 2
#define CT 4096
#define CE 768
#define CH 12
#define CD 64
#define GM (CB * CT)  // 8192 rows for the projection GEMMs

typedef __attribute__((ext_vector_type(16))) __bf16 v16bf;
typedef __attribute__((ext_vector_type(8)))  float  v8f;

union AFragU { uint32_t u[8]; v16bf v; };

__device__ inline uint16_t f2bf(float x) {
  uint32_t u = __float_as_uint(x);
  u += 0x7FFFu + ((u >> 16) & 1u);
  return (uint16_t)(u >> 16);
}

// A-operand (16x32, 16-bit): lane m = lane&15 holds row m.
// lanes 0-15: K = 0..7 then 16..23 ; lanes 16-31: K = 8..15 then 24..31
// base points at element [row0][k0] of a row-major bf16 array with leading dim ld.
__device__ inline v16bf load_frag_a(const uint16_t* base, int ld, int lane) {
  const uint16_t* p = base + (size_t)(lane & 15) * ld + ((lane >> 4) << 3);
  AFragU f;
#pragma unroll
  for (int q = 0; q < 2; ++q)
#pragma unroll
    for (int j = 0; j < 4; ++j)
      f.u[q * 4 + j] = *(const uint32_t*)(p + q * 16 + 2 * j);
  return f.v;
}

// B-operand (32x16, 16-bit): lane n = lane&15 holds column n.
// lanes 0-15: K = 0..7 then 16..23 ; lanes 16-31: K = 8..15 then 24..31
// base points at element [n0][k0] of B^T stored row-major (row = n, ld along k).
__device__ inline v16bf load_frag_b(const uint16_t* base, int ld, int lane) {
  const uint16_t* p = base + (size_t)(lane & 15) * ld + ((lane >> 4) << 3);
  AFragU f;
#pragma unroll
  for (int j = 0; j < 8; ++j)
    f.u[j] = *(const uint32_t*)(p + (j >> 2) * 16 + 2 * (j & 3));
  return f.v;
}

// ---------------------------------------------------------------- converts --
__global__ void convert_bf16_kernel(const float* __restrict__ src,
                                    uint16_t* __restrict__ dst, size_t n) {
  size_t stride = (size_t)gridDim.x * blockDim.x;
  for (size_t i = (size_t)blockIdx.x * blockDim.x + threadIdx.x; i < n; i += stride)
    dst[i] = f2bf(src[i]);
}

// src: (K,N) fp32 row-major ; dst: (N,K) bf16 row-major (i.e. W^T)
__global__ void transpose_bf16_kernel(const float* __restrict__ src,
                                      uint16_t* __restrict__ dst, int K, int N) {
  size_t total = (size_t)K * N;
  size_t stride = (size_t)gridDim.x * blockDim.x;
  for (size_t i = (size_t)blockIdx.x * blockDim.x + threadIdx.x; i < total; i += stride) {
    size_t nn = i / (size_t)K;
    size_t kk = i % (size_t)K;
    dst[i] = f2bf(src[kk * (size_t)N + nn]);
  }
}

// ---------------------------------------------------------------- GEMM ------
// C[m][e] = sum_k A[m][k] * W[k][e] + bias[e], M=GM, N=K=CE.
// Register-blocked: each wave computes a 64x64 output macro-tile (4x4 WMMA
// tiles, 16 v8f accumulators). Per 32-wide k-step: 4 A-frags + 4 B-frags,
// 16 back-to-back v_wmma -> 4x the FLOP/byte of a 16x16-per-wave design.
// A: (GM,CE) bf16 row-major. BT: W^T (CE,CE) bf16 row-major.
// mode 0: write bf16 to (B,H,T,D)      (Q or K heads layout)
// mode 1: write bf16 to (B,H,D,T)      (V transposed for P*V B-operand)
// mode 2: write fp32 to (B,T,E)        (final output)
__global__ void __launch_bounds__(256)
gemm_bias_kernel(const uint16_t* __restrict__ A, const uint16_t* __restrict__ BT,
                 const float* __restrict__ bias, void* __restrict__ outp, int mode) {
  const int lane = threadIdx.x & 31;
  const int wid  = threadIdx.x >> 5;
  const int tile = blockIdx.x * 8 + wid;   // 64x64 macro-tile id
  const int NT   = CE / 64;                // 12 macro cols
  const int nt   = tile % NT;
  const int mt   = tile / NT;              // 0..127

  v8f acc[4][4];
#pragma unroll
  for (int mi = 0; mi < 4; ++mi)
#pragma unroll
    for (int ni = 0; ni < 4; ++ni)
#pragma unroll
      for (int i = 0; i < 8; ++i) acc[mi][ni][i] = 0.f;

  const uint16_t* arow = A + (size_t)(mt * 64) * CE;
  const uint16_t* brow = BT + (size_t)(nt * 64) * CE;
  for (int k0 = 0; k0 < CE; k0 += 32) {
    v16bf a[4], b[4];
#pragma unroll
    for (int mi = 0; mi < 4; ++mi)
      a[mi] = load_frag_a(arow + (size_t)(mi * 16) * CE + k0, CE, lane);
#pragma unroll
    for (int ni = 0; ni < 4; ++ni)
      b[ni] = load_frag_b(brow + (size_t)(ni * 16) * CE + k0, CE, lane);
#pragma unroll
    for (int mi = 0; mi < 4; ++mi)
#pragma unroll
      for (int ni = 0; ni < 4; ++ni)
        acc[mi][ni] = __builtin_amdgcn_wmma_f32_16x16x32_bf16(
            false, a[mi], false, b[ni], (short)0, acc[mi][ni], false, false);
  }

  const int g = lane >> 4;
  const int n = lane & 15;

#pragma unroll
  for (int ni = 0; ni < 4; ++ni) {
    const int e  = nt * 64 + ni * 16 + n;
    const float bv = bias[e];
    if (mode == 2) {
      float* out = (float*)outp;
#pragma unroll
      for (int mi = 0; mi < 4; ++mi)
#pragma unroll
        for (int i = 0; i < 8; ++i) {
          int m = mt * 64 + mi * 16 + g * 8 + i;
          out[(size_t)m * CE + e] = acc[mi][ni][i] + bv;
        }
    } else if (mode == 0) {
      uint16_t* out = (uint16_t*)outp;
      const int h = e >> 6, d = e & 63;
#pragma unroll
      for (int mi = 0; mi < 4; ++mi)
#pragma unroll
        for (int i = 0; i < 8; ++i) {
          int m  = mt * 64 + mi * 16 + g * 8 + i;
          int bb = m >> 12;
          int t  = m & (CT - 1);
          out[(((size_t)bb * CH + h) * CT + t) * CD + d] = f2bf(acc[mi][ni][i] + bv);
        }
    } else {  // mode 1: V transposed; 8 rows are contiguous in t -> one b128 store
      uint16_t* out = (uint16_t*)outp;
      const int h = e >> 6, d = e & 63;
#pragma unroll
      for (int mi = 0; mi < 4; ++mi) {
        int m0 = mt * 64 + mi * 16 + g * 8;     // multiple of 8; batch-uniform
        int bb = m0 >> 12;
        int t0 = m0 & (CT - 1);
        uint16_t tmp[8];
#pragma unroll
        for (int i = 0; i < 8; ++i) tmp[i] = f2bf(acc[mi][ni][i] + bv);
        size_t addr = (((size_t)bb * CH + h) * CD + d) * CT + t0;
        *(uint4*)(out + addr) = *(const uint4*)tmp;
      }
    }
  }
}

// ---------------------------------------------------------- flash attention --
// Q,K: (B,H,T,D) bf16.  Vt: (B,H,D,T) bf16.  O: (B,T,E) bf16.
// grid = (B*H, T/128), block = 128 (4 waves). Each wave owns 32 query rows
// (two 16-row Q tiles) so every K/V block is reused twice.
__global__ void __launch_bounds__(128)
flash_attn_kernel(const uint16_t* __restrict__ Q, const uint16_t* __restrict__ K,
                  const uint16_t* __restrict__ Vt, uint16_t* __restrict__ O) {
  __shared__ uint16_t pbuf[4][2][16][32];

  const int lane = threadIdx.x & 31;
  const int w    = threadIdx.x >> 5;
  const int bh   = blockIdx.x;
  const int b    = bh / CH;
  const int h    = bh % CH;
  const int q0   = blockIdx.y * 128 + w * 32;  // first query row for this wave
  const int g    = lane >> 4;
  const int n    = lane & 15;

  v16bf qa[2][2];
#pragma unroll
  for (int qt = 0; qt < 2; ++qt) {
    const uint16_t* Qb = Q + ((size_t)bh * CT + q0 + qt * 16) * CD;
    qa[qt][0] = load_frag_a(Qb + 0, CD, lane);
    qa[qt][1] = load_frag_a(Qb + 32, CD, lane);
  }

  float m[2][8], l[2][8];
  v8f acc[2][4];
#pragma unroll
  for (int qt = 0; qt < 2; ++qt) {
#pragma unroll
    for (int i = 0; i < 8; ++i) { m[qt][i] = -1e30f; l[qt][i] = 0.f; }
#pragma unroll
    for (int db = 0; db < 4; ++db)
#pragma unroll
      for (int i = 0; i < 8; ++i) acc[qt][db][i] = 0.f;
  }

  for (int s0 = 0; s0 < q0 + 32; s0 += 32) {
    // ---- S = (Q K^T) * scale for key columns [s0, s0+32), causal-masked ----
    float s[2][2][8];  // [qtile][stile][row-elem]
#pragma unroll
    for (int tt = 0; tt < 2; ++tt) {
      const uint16_t* Kb = K + ((size_t)bh * CT + s0 + tt * 16) * CD;
      v16bf kb0 = load_frag_b(Kb + 0, CD, lane);
      v16bf kb1 = load_frag_b(Kb + 32, CD, lane);
      const int col = s0 + tt * 16 + n;
#pragma unroll
      for (int qt = 0; qt < 2; ++qt) {
        v8f sa = {0.f, 0.f, 0.f, 0.f, 0.f, 0.f, 0.f, 0.f};
        sa = __builtin_amdgcn_wmma_f32_16x16x32_bf16(false, qa[qt][0], false, kb0,
                                                     (short)0, sa, false, false);
        sa = __builtin_amdgcn_wmma_f32_16x16x32_bf16(false, qa[qt][1], false, kb1,
                                                     (short)0, sa, false, false);
#pragma unroll
        for (int i = 0; i < 8; ++i) {
          int row = q0 + qt * 16 + g * 8 + i;
          float v = sa[i] * 0.125f;          // 1/sqrt(64)
          s[qt][tt][i] = (col <= row) ? v : -1e30f;
        }
      }
    }

    // ---- online softmax per q tile (rows per-element, cols per-lane) ----
#pragma unroll
    for (int qt = 0; qt < 2; ++qt) {
#pragma unroll
      for (int i = 0; i < 8; ++i) {
        float mb = fmaxf(s[qt][0][i], s[qt][1][i]);
        mb = fmaxf(mb, __shfl_xor(mb, 1, 32));
        mb = fmaxf(mb, __shfl_xor(mb, 2, 32));
        mb = fmaxf(mb, __shfl_xor(mb, 4, 32));
        mb = fmaxf(mb, __shfl_xor(mb, 8, 32));
        float mn = fmaxf(m[qt][i], mb);
        float sc = __expf(m[qt][i] - mn);
        float p0 = __expf(s[qt][0][i] - mn);
        float p1 = __expf(s[qt][1][i] - mn);
        float rs = p0 + p1;
        rs += __shfl_xor(rs, 1, 32);
        rs += __shfl_xor(rs, 2, 32);
        rs += __shfl_xor(rs, 4, 32);
        rs += __shfl_xor(rs, 8, 32);
        l[qt][i] = l[qt][i] * sc + rs;
        m[qt][i] = mn;
#pragma unroll
        for (int db = 0; db < 4; ++db) acc[qt][db][i] *= sc;
        pbuf[w][qt][g * 8 + i][n]      = f2bf(p0);
        pbuf[w][qt][g * 8 + i][16 + n] = f2bf(p1);
      }
    }
    asm volatile("s_wait_dscnt 0" ::: "memory");  // cross-lane LDS RAW (per-wave)

    // ---- O += P * V  (V frags loaded once, reused by both q tiles) ----
    v16bf vb[4];
#pragma unroll
    for (int db = 0; db < 4; ++db)
      vb[db] = load_frag_b(Vt + ((size_t)bh * CD + db * 16) * CT + s0, CT, lane);
#pragma unroll
    for (int qt = 0; qt < 2; ++qt) {
      v16bf pa = load_frag_a(&pbuf[w][qt][0][0], 32, lane);
#pragma unroll
      for (int db = 0; db < 4; ++db)
        acc[qt][db] = __builtin_amdgcn_wmma_f32_16x16x32_bf16(
            false, pa, false, vb[db], (short)0, acc[qt][db], false, false);
    }
    asm volatile("s_wait_dscnt 0" ::: "memory");  // P reads done before next store
  }

  // ---- epilogue: O / l  -> attn (B,T,E) bf16 ----
#pragma unroll
  for (int qt = 0; qt < 2; ++qt)
#pragma unroll
    for (int db = 0; db < 4; ++db)
#pragma unroll
      for (int i = 0; i < 8; ++i) {
        int t = q0 + qt * 16 + g * 8 + i;
        int e = h * 64 + db * 16 + n;
        O[((size_t)b * CT + t) * CE + e] = f2bf(acc[qt][db][i] / l[qt][i]);
      }
}

// ---------------------------------------------------------------- launch ----
extern "C" void kernel_launch(void* const* d_in, const int* in_sizes, int n_in,
                              void* d_out, int out_size, void* d_ws, size_t ws_size,
                              hipStream_t stream) {
  const float* x  = (const float*)d_in[0];
  const float* wq = (const float*)d_in[1];
  const float* bq = (const float*)d_in[2];
  const float* wk = (const float*)d_in[3];
  const float* bk = (const float*)d_in[4];
  const float* wv = (const float*)d_in[5];
  const float* bv = (const float*)d_in[6];
  const float* wo = (const float*)d_in[7];
  const float* bo = (const float*)d_in[8];

  const size_t actBytes = (size_t)GM * CE * sizeof(uint16_t);  // 12.6 MB
  const size_t wBytes   = (size_t)CE * CE * sizeof(uint16_t);  // 1.18 MB

  char* p = (char*)d_ws;
  uint16_t* xb    = (uint16_t*)p; p += actBytes;
  uint16_t* wqT   = (uint16_t*)p; p += wBytes;
  uint16_t* wkT   = (uint16_t*)p; p += wBytes;
  uint16_t* wvT   = (uint16_t*)p; p += wBytes;
  uint16_t* woT   = (uint16_t*)p; p += wBytes;
  uint16_t* qb    = (uint16_t*)p; p += actBytes;
  uint16_t* kb    = (uint16_t*)p; p += actBytes;
  uint16_t* vT    = (uint16_t*)p; p += actBytes;
  uint16_t* attnb = (uint16_t*)p; p += actBytes;

  // 1) fp32 -> bf16 conversions (weights transposed to W^T for B-operand loads)
  convert_bf16_kernel<<<1024, 256, 0, stream>>>(x, xb, (size_t)GM * CE);
  transpose_bf16_kernel<<<512, 256, 0, stream>>>(wq, wqT, CE, CE);
  transpose_bf16_kernel<<<512, 256, 0, stream>>>(wk, wkT, CE, CE);
  transpose_bf16_kernel<<<512, 256, 0, stream>>>(wv, wvT, CE, CE);
  transpose_bf16_kernel<<<512, 256, 0, stream>>>(wo, woT, CE, CE);

  // 2) QKV projections (WMMA bf16, fp32 accumulate), 64x64 per wave
  const int gemmBlocks = (GM / 64) * (CE / 64) / 8;  // 192
  gemm_bias_kernel<<<gemmBlocks, 256, 0, stream>>>(xb, wqT, bq, qb, 0);
  gemm_bias_kernel<<<gemmBlocks, 256, 0, stream>>>(xb, wkT, bk, kb, 0);
  gemm_bias_kernel<<<gemmBlocks, 256, 0, stream>>>(xb, wvT, bv, vT, 1);

  // 3) causal flash attention (32 query rows per wave)
  flash_attn_kernel<<<dim3(CB * CH, CT / 128), 128, 0, stream>>>(qb, kb, vT, attnb);

  // 4) output projection -> fp32 d_out (B,T,E)
  gemm_bias_kernel<<<gemmBlocks, 256, 0, stream>>>(attnb, woT, bo, d_out, 2);
}
